// LinearAttention_7876970021338
// MI455X (gfx1250) — matrix operand order repaired and checked by reading, not compiled, not run on hardware
//
#include <hip/hip_runtime.h>
#include <hip/hip_bf16.h>

// ---------------------------------------------------------------------------
// Linear attention on MI455X (gfx1250):
//  - all contractions on v_wmma_f32_16x16x32_bf16 (f32 accumulate)
//  - double-buffered LDS stages fed by global_load_async_to_lds_b128 (ASYNCcnt)
//    when available, sync global->VGPR->LDS otherwise
// ---------------------------------------------------------------------------

#define Bc 8
#define Nc 4096
#define Dc 1024
#define EPSc 1e-6f

#define LDA 40    // padded LDS row stride (bf16 elems) for 32-wide K slabs
#define LDB 40
#define SA_STRIDE (64 * LDA)    // one A stage
#define SB_STRIDE (256 * LDB)   // one B stage

typedef __attribute__((ext_vector_type(16))) __bf16 bf16x16;
typedef __attribute__((ext_vector_type(8)))  __bf16 bf16x8;
typedef __attribute__((ext_vector_type(4)))  __bf16 bf16x4;
typedef __attribute__((ext_vector_type(8)))  float  f32x8;
typedef __attribute__((ext_vector_type(4)))  int    i32x4;

#if defined(__AMDGCN__) && defined(__has_builtin)
#if __has_builtin(__builtin_amdgcn_global_load_async_to_lds_b128) && \
    __has_builtin(__builtin_amdgcn_s_wait_asynccnt)
#define USE_ASYNC_LDS 1
#endif
#endif
#ifndef USE_ASYNC_LDS
#define USE_ASYNC_LDS 0
#endif

__device__ __forceinline__ __bf16 f2bf(float f) {
    unsigned u = __float_as_uint(f);
    u += 0x7FFFu + ((u >> 16) & 1u);            // round-to-nearest-even
    union { unsigned short s; __bf16 b; } cv;
    cv.s = (unsigned short)(u >> 16);
    return cv.b;
}

__device__ __forceinline__ float bf2f(__bf16 b) {
    union { __bf16 b; unsigned short s; } cv;
    cv.b = b;
    return __uint_as_float(((unsigned)cv.s) << 16);
}

// 16B global -> LDS transfer (async DMA path or sync fallback).
// Builtin signature (from toolchain diagnostic): param0 = int4 addrspace(1)*
// (prints as "int4 __device__*" in HIP), param1 = LDS side, then imm offset,
// imm cpol.
__device__ __forceinline__ void g2l16(const __bf16* g, __bf16* l) {
#if USE_ASYNC_LDS
    __builtin_amdgcn_global_load_async_to_lds_b128(
        (__attribute__((address_space(1))) i32x4*)(g),
        (__attribute__((address_space(3))) i32x4*)(l),
        0, 0);
#else
    *(bf16x8*)l = *(const bf16x8*)g;
#endif
}

// ---- WMMA fragment loads from LDS (contiguous 16B reads -> ds_load_b128) ----
// A 16x32 bf16: lane m = lane&15, half = lane>>4; elems 0..7 <- K = half*8+e,
// elems 8..15 <- K = 16+half*8+e.
__device__ __forceinline__ bf16x16 frag_a(const __bf16* sA, int mt, int lane) {
    const int m = lane & 15, h = lane >> 4;
    const __bf16* r = sA + (mt * 16 + m) * LDA + h * 8;
    union { bf16x16 v; bf16x8 p[2]; } u;
    u.p[0] = *(const bf16x8*)(r);
    u.p[1] = *(const bf16x8*)(r + 16);
    return u.v;
}
// B 32x16 bf16 read from B^T row-major tile: lane n = lane&15; elems 0..7 <- K = half*8+e,
// elems 8..15 <- K = 16+half*8+e.
__device__ __forceinline__ bf16x16 frag_b(const __bf16* sB, int sub, int lane) {
    const int n = lane & 15, h = lane >> 4;
    const __bf16* r = sB + (sub * 16 + n) * LDB + h * 8;
    union { bf16x16 v; bf16x8 p[2]; } u;
    u.p[0] = *(const bf16x8*)(r);
    u.p[1] = *(const bf16x8*)(r + 16);
    return u.v;
}

// One K-stage of cooperative loads: A 64x32 (256 chunks), B^T 256x32 (1024
// chunks). 256 threads -> 5 x 16B transfers each.
__device__ __forceinline__ void stage_load(
    const __bf16* __restrict__ gA, const __bf16* __restrict__ gB,
    int lda, int ldb, int k0, __bf16* dA, __bf16* dB, int r, int c)
{
    g2l16(gA + (size_t)r * lda + k0 + c * 8, dA + r * LDA + c * 8);
#pragma unroll
    for (int j = 0; j < 4; ++j)
        g2l16(gB + (size_t)(r + 64 * j) * ldb + k0 + c * 8,
              dB + (r + 64 * j) * LDB + c * 8);
    // warm L2 for the slab after next (speculative, no counter cost)
    __builtin_prefetch(gA + (size_t)r * lda + k0 + 64 + c * 8, 0, 1);
    __builtin_prefetch(gB + (size_t)r * ldb + k0 + 64 + c * 8, 0, 1);
}

// ---- Shared GEMM mainloop: C(64x256) += A(64xK) @ B^T(256xK)^T --------------
// Wave tile 64x32: 4 M-frags x 2 N-frags -> 8 WMMAs per 12 ds_load_b128.
// Double-buffered LDS; stage s+1 issued before waiting/consuming stage s.
__device__ __forceinline__ void gemm_mainloop(
    const __bf16* __restrict__ gA, const __bf16* __restrict__ gB,
    int lda, int ldb, int ktot,
    __bf16* sA, __bf16* sB, f32x8 acc[4][2], int tid)
{
    const int r = tid >> 2, c = tid & 3;
    const int lane = tid & 31, wave = tid >> 5;
    const int nstage = ktot >> 5;

    stage_load(gA, gB, lda, ldb, 0, sA, sB, r, c);
    for (int s = 0; s < nstage; ++s) {
        __bf16* cA = sA + (s & 1) * SA_STRIDE;
        __bf16* cB = sB + (s & 1) * SB_STRIDE;
        const bool more = (s + 1) < nstage;
        if (more)
            stage_load(gA, gB, lda, ldb, (s + 1) << 5,
                       sA + ((s + 1) & 1) * SA_STRIDE,
                       sB + ((s + 1) & 1) * SB_STRIDE, r, c);
#if USE_ASYNC_LDS
        // 5 async ops per stage, in-order completion: <=5 outstanding means the
        // stage we are about to consume has fully landed in LDS.
        if (more) __builtin_amdgcn_s_wait_asynccnt(5);
        else      __builtin_amdgcn_s_wait_asynccnt(0);
#endif
        __syncthreads();
        bf16x16 bf0 = frag_b(cB, wave * 2 + 0, lane);
        bf16x16 bf1 = frag_b(cB, wave * 2 + 1, lane);
#pragma unroll
        for (int mt = 0; mt < 4; ++mt) {
            bf16x16 af = frag_a(cA, mt, lane);
            acc[mt][0] = __builtin_amdgcn_wmma_f32_16x16x32_bf16(
                false, af, false, bf0, (short)0, acc[mt][0], false, false);
            acc[mt][1] = __builtin_amdgcn_wmma_f32_16x16x32_bf16(
                false, af, false, bf1, (short)0, acc[mt][1], false, false);
        }
        __syncthreads();
    }
}

// ---------------------------------------------------------------------------
// K0: fp32 -> bf16 conversion (4 elems/thread)
// ---------------------------------------------------------------------------
__global__ void cvt_bf16_kernel(const float* __restrict__ src, __bf16* __restrict__ dst, int n) {
    int i = (blockIdx.x * 256 + threadIdx.x) * 4;
    if (i + 3 < n) {
        float4 f = *(const float4*)(src + i);
        union { bf16x4 v; __bf16 e[4]; } o;
        o.e[0] = f2bf(f.x); o.e[1] = f2bf(f.y); o.e[2] = f2bf(f.z); o.e[3] = f2bf(f.w);
        *(bf16x4*)(dst + i) = o.v;
    }
}

// ---------------------------------------------------------------------------
// K1: qkv = x @ W_qkv^T, fused relu+eps; routes q (row-major), kT, vT (transposed)
// grid (3072/256, 4096/64, B), block 256
// ---------------------------------------------------------------------------
__global__ void qkv_gemm_kernel(const __bf16* __restrict__ xb, const __bf16* __restrict__ wb,
                                __bf16* __restrict__ qb, __bf16* __restrict__ kT,
                                __bf16* __restrict__ vT)
{
    __shared__ __bf16 sA[2 * SA_STRIDE];
    __shared__ __bf16 sB[2 * SB_STRIDE];
    const int b = blockIdx.z;
    const int mBase = blockIdx.y * 64;     // token tile
    const int nBase = blockIdx.x * 256;    // feature tile in [0,3072)
    const int tid = threadIdx.x;
    const __bf16* gA = xb + ((size_t)b * Nc + mBase) * Dc;
    const __bf16* gB = wb + (size_t)nBase * Dc;
    f32x8 acc[4][2] = {};
    gemm_mainloop(gA, gB, Dc, Dc, Dc, sA, sB, acc, tid);

    const int lane = tid & 31, wave = tid >> 5, half = lane >> 4;
#pragma unroll
    for (int nt = 0; nt < 2; ++nt) {
        const int nGlob = nBase + wave * 32 + nt * 16 + (lane & 15);
        const int seg = nGlob >> 10;       // 0:q 1:k 2:v (32-wide window never straddles)
        const int f = nGlob & 1023;
#pragma unroll
        for (int mt = 0; mt < 4; ++mt) {
            if (seg == 0) {
#pragma unroll
                for (int v = 0; v < 8; ++v) {
                    int token = mBase + mt * 16 + half * 8 + v;
                    float val = fmaxf(acc[mt][nt][v], 0.f) + EPSc;
                    qb[((size_t)b * Nc + token) * Dc + f] = f2bf(val);
                }
            } else {
                union { bf16x8 v8; __bf16 e[8]; } st;
#pragma unroll
                for (int v = 0; v < 8; ++v) {
                    float val = acc[mt][nt][v];
                    if (seg == 1) val = fmaxf(val, 0.f) + EPSc;
                    st.e[v] = f2bf(val);
                }
                // transposed store: M contiguous across acc VGPRs -> one 16B store
                __bf16* dst = (seg == 1 ? kT : vT)
                            + ((size_t)b * Dc + f) * Nc + mBase + mt * 16 + half * 8;
                *(bf16x8*)dst = st.v8;
            }
        }
    }
}

// ---------------------------------------------------------------------------
// K2: ksum[b][d] = sum_n kT[b][d][n]  (wave per row, 16B/lane)
// ---------------------------------------------------------------------------
__global__ void ksum_kernel(const __bf16* __restrict__ kT, float* __restrict__ ksum) {
    const int row = blockIdx.x * 8 + (threadIdx.x >> 5);   // b*D + d
    const int lane = threadIdx.x & 31;
    const __bf16* src = kT + (size_t)row * Nc;
    float s = 0.f;
    for (int i = lane * 8; i < Nc; i += 32 * 8) {
        bf16x8 v = *(const bf16x8*)(src + i);
#pragma unroll
        for (int j = 0; j < 8; ++j) s += bf2f(v[j]);
    }
#pragma unroll
    for (int m = 16; m; m >>= 1) s += __shfl_xor(s, m, 32);
    if (lane == 0) ksum[row] = s;
}

// ---------------------------------------------------------------------------
// K3: z[b][n] = 1/(q[b][n]·ksum[b] + eps)  (wave per token)
// ---------------------------------------------------------------------------
__global__ void z_kernel(const __bf16* __restrict__ qb, const float* __restrict__ ksum,
                         float* __restrict__ z) {
    const int token = blockIdx.x * 8 + (threadIdx.x >> 5); // b*N + n
    const int b = token >> 12;                             // N = 4096
    const int lane = threadIdx.x & 31;
    const __bf16* qrow = qb + (size_t)token * Dc;
    const float* ks = ksum + (size_t)b * Dc;
    float s = 0.f;
    for (int i = lane * 8; i < Dc; i += 32 * 8) {
        bf16x8 qv = *(const bf16x8*)(qrow + i);
#pragma unroll
        for (int j = 0; j < 8; ++j) s += bf2f(qv[j]) * ks[i + j];
    }
#pragma unroll
    for (int m = 16; m; m >>= 1) s += __shfl_xor(s, m, 32);
    if (lane == 0) z[token] = 1.f / (s + EPSc);
}

// ---------------------------------------------------------------------------
// K4: kv[d][e] = sum_n k[n][d] v[n][e]; stored transposed as kvT[b][e][d] bf16
// grid (1024/256, 1024/64, B)
// ---------------------------------------------------------------------------
__global__ void kv_gemm_kernel(const __bf16* __restrict__ kT, const __bf16* __restrict__ vT,
                               __bf16* __restrict__ kvT)
{
    __shared__ __bf16 sA[2 * SA_STRIDE];
    __shared__ __bf16 sB[2 * SB_STRIDE];
    const int b = blockIdx.z;
    const int mBase = blockIdx.y * 64;     // d tile
    const int nBase = blockIdx.x * 256;    // e tile
    const int tid = threadIdx.x;
    const __bf16* gA = kT + ((size_t)b * Dc + mBase) * Nc;
    const __bf16* gB = vT + ((size_t)b * Dc + nBase) * Nc;
    f32x8 acc[4][2] = {};
    gemm_mainloop(gA, gB, Nc, Nc, Nc, sA, sB, acc, tid);

    const int lane = tid & 31, wave = tid >> 5, half = lane >> 4;
#pragma unroll
    for (int nt = 0; nt < 2; ++nt) {
        const int eGlob = nBase + wave * 32 + nt * 16 + (lane & 15);
#pragma unroll
        for (int mt = 0; mt < 4; ++mt) {
            union { bf16x8 v8; __bf16 e[8]; } st;
#pragma unroll
            for (int v = 0; v < 8; ++v) st.e[v] = f2bf(acc[mt][nt][v]);
            __bf16* dst = kvT + ((size_t)b * Dc + eGlob) * Dc + mBase + mt * 16 + half * 8;
            *(bf16x8*)dst = st.v8;
        }
    }
}

// ---------------------------------------------------------------------------
// K5: attn[n][e] = z[n] * sum_d q[n][d] kv[d][e]  -> bf16 row-major
// grid (1024/256, 4096/64, B)
// ---------------------------------------------------------------------------
__global__ void attn_gemm_kernel(const __bf16* __restrict__ qb, const __bf16* __restrict__ kvT,
                                 const float* __restrict__ z, __bf16* __restrict__ attnb)
{
    __shared__ __bf16 sA[2 * SA_STRIDE];
    __shared__ __bf16 sB[2 * SB_STRIDE];
    const int b = blockIdx.z;
    const int mBase = blockIdx.y * 64;
    const int nBase = blockIdx.x * 256;
    const int tid = threadIdx.x;
    const __bf16* gA = qb + ((size_t)b * Nc + mBase) * Dc;
    const __bf16* gB = kvT + ((size_t)b * Dc + nBase) * Dc;
    f32x8 acc[4][2] = {};
    gemm_mainloop(gA, gB, Dc, Dc, Dc, sA, sB, acc, tid);

    const int lane = tid & 31, wave = tid >> 5, half = lane >> 4;
#pragma unroll
    for (int mt = 0; mt < 4; ++mt) {
#pragma unroll
        for (int v = 0; v < 8; ++v) {
            const int token = mBase + mt * 16 + half * 8 + v;
            const float zz = z[(size_t)b * Nc + token];
#pragma unroll
            for (int nt = 0; nt < 2; ++nt) {
                const int eGlob = nBase + wave * 32 + nt * 16 + (lane & 15);
                attnb[((size_t)b * Nc + token) * Dc + eGlob] = f2bf(acc[mt][nt][v] * zz);
            }
        }
    }
}

// ---------------------------------------------------------------------------
// K6: out = attn @ W_out^T + b_out  (fp32 output)
// grid (1024/256, 4096/64, B)
// ---------------------------------------------------------------------------
__global__ void out_gemm_kernel(const __bf16* __restrict__ attnb, const __bf16* __restrict__ wob,
                                const float* __restrict__ bout, float* __restrict__ out)
{
    __shared__ __bf16 sA[2 * SA_STRIDE];
    __shared__ __bf16 sB[2 * SB_STRIDE];
    const int b = blockIdx.z;
    const int mBase = blockIdx.y * 64;
    const int nBase = blockIdx.x * 256;
    const int tid = threadIdx.x;
    const __bf16* gA = attnb + ((size_t)b * Nc + mBase) * Dc;
    const __bf16* gB = wob + (size_t)nBase * Dc;
    f32x8 acc[4][2] = {};
    gemm_mainloop(gA, gB, Dc, Dc, Dc, sA, sB, acc, tid);

    const int lane = tid & 31, wave = tid >> 5, half = lane >> 4;
#pragma unroll
    for (int nt = 0; nt < 2; ++nt) {
        const int fGlob = nBase + wave * 32 + nt * 16 + (lane & 15);
        const float bias = bout[fGlob];
#pragma unroll
        for (int mt = 0; mt < 4; ++mt) {
#pragma unroll
            for (int v = 0; v < 8; ++v) {
                const int token = mBase + mt * 16 + half * 8 + v;
                out[((size_t)b * Nc + token) * Dc + fGlob] = acc[mt][nt][v] + bias;
            }
        }
    }
}

// ---------------------------------------------------------------------------
extern "C" void kernel_launch(void* const* d_in, const int* in_sizes, int n_in,
                              void* d_out, int out_size, void* d_ws, size_t ws_size,
                              hipStream_t stream) {
    const float* x    = (const float*)d_in[0];   // [B,N,D]
    const float* Wqkv = (const float*)d_in[1];   // [3D,D]
    const float* Wout = (const float*)d_in[2];   // [D,D]
    const float* bout = (const float*)d_in[3];   // [D]
    float* out = (float*)d_out;                  // [B,N,D] fp32

    // workspace carve (256B aligned regions)
    char* p = (char*)d_ws;
    auto carve = [&](size_t bytes) -> void* {
        void* r = (void*)p;
        p += (bytes + 255) & ~(size_t)255;
        return r;
    };
    const size_t BND = (size_t)Bc * Nc * Dc;
    __bf16* xb    = (__bf16*)carve(BND * 2);                 // x bf16 (reused as attn later)
    __bf16* wqkvb = (__bf16*)carve((size_t)3 * Dc * Dc * 2);
    __bf16* woutb = (__bf16*)carve((size_t)Dc * Dc * 2);
    __bf16* qb    = (__bf16*)carve(BND * 2);                 // q row-major
    __bf16* kTb   = (__bf16*)carve(BND * 2);                 // k transposed [B,D,N]
    __bf16* vTb   = (__bf16*)carve(BND * 2);                 // v transposed [B,D,N]
    __bf16* kvTb  = (__bf16*)carve((size_t)Bc * Dc * Dc * 2);// kv transposed [B,D(e),D(d)]
    float*  ksum  = (float*)carve((size_t)Bc * Dc * 4);
    float*  zbuf  = (float*)carve((size_t)Bc * Nc * 4);
    __bf16* attnb = xb;  // x dead after K1 -> reuse for attn

    // K0: conversions
    cvt_bf16_kernel<<<(unsigned)(BND / 1024), 256, 0, stream>>>(x, xb, (int)BND);
    cvt_bf16_kernel<<<(3 * Dc * Dc) / 1024, 256, 0, stream>>>(Wqkv, wqkvb, 3 * Dc * Dc);
    cvt_bf16_kernel<<<(Dc * Dc) / 1024, 256, 0, stream>>>(Wout, woutb, Dc * Dc);

    // K1: qkv projection + relu/eps + layout routing
    qkv_gemm_kernel<<<dim3(3 * Dc / 256, Nc / 64, Bc), 256, 0, stream>>>(xb, wqkvb, qb, kTb, vTb);

    // K2/K3: ksum and z
    ksum_kernel<<<(Bc * Dc) / 8, 256, 0, stream>>>(kTb, ksum);
    z_kernel<<<(Bc * Nc) / 8, 256, 0, stream>>>(qb, ksum, zbuf);

    // K4: kv = k^T v  (stored transposed)
    kv_gemm_kernel<<<dim3(Dc / 256, Dc / 64, Bc), 256, 0, stream>>>(kTb, vTb, kvTb);

    // K5: attn = (q @ kv) * z
    attn_gemm_kernel<<<dim3(Dc / 256, Nc / 64, Bc), 256, 0, stream>>>(qb, kvTb, zbuf, attnb);

    // K6: out = attn @ W_out^T + b_out
    out_gemm_kernel<<<dim3(Dc / 256, Nc / 64, Bc), 256, 0, stream>>>(attnb, woutb, bout, out);
}
